// MultiView_GNN_44968307589650
// MI455X (gfx1250) — matrix operand order repaired and checked by reading, not compiled, not run on hardware
//
#include <hip/hip_runtime.h>
#include <math.h>

#define NND 3072
#define HD  64
#define NV  4
#define ALPHA_F 0.8f
#define BETA_F  0.5f
#define LRELU_F 0.2f

typedef __attribute__((ext_vector_type(16))) __bf16        v16bf;
typedef __attribute__((ext_vector_type(8)))  float         v8f;
typedef __attribute__((ext_vector_type(4)))  float         f32x4;
typedef __attribute__((ext_vector_type(4)))  unsigned int  u32x4;

union FragU { unsigned int w[8]; u32x4 q[2]; v16bf bf; };

static __device__ __forceinline__ unsigned pack_bf2(float lo, float hi) {
  return (__float_as_uint(lo) >> 16) | (__float_as_uint(hi) & 0xffff0000u);
}
static __device__ __forceinline__ unsigned short f2bf(float x) {
  return (unsigned short)(__float_as_uint(x) >> 16);
}
static __device__ __forceinline__ v8f wmma_bf16(v16bf a, v16bf b, v8f c) {
  return __builtin_amdgcn_wmma_f32_16x16x32_bf16(false, a, false, b, (short)0, c,
                                                 false, false);
}

// A fragment (16x32 bf16) from row-major bf16, row stride ld elems.
// lane l: row = l&15; K = {h..h+7, 16+h..16+h+7}, h = 8*(l>=16).
static __device__ __forceinline__ v16bf load_a_bf16(const unsigned short* base,
                                                    int ld, int lane) {
  int r = lane & 15, h = (lane >> 4) << 3;
  FragU u;
  u.q[0] = *(const u32x4*)(base + r * ld + h);
  u.q[1] = *(const u32x4*)(base + r * ld + 16 + h);
  return u.bf;
}
// A fragment from f32 row-major (convert to bf16 on the fly)
static __device__ __forceinline__ v16bf load_a_f32(const float* base, int ld,
                                                   int lane) {
  int r = lane & 15, h = (lane >> 4) << 3;
  const float* p = base + r * ld;
  FragU u;
#pragma unroll
  for (int i = 0; i < 4; ++i) u.w[i]     = pack_bf2(p[h + 2*i],      p[h + 2*i + 1]);
#pragma unroll
  for (int i = 0; i < 4; ++i) u.w[4 + i] = pack_bf2(p[16 + h + 2*i], p[16 + h + 2*i + 1]);
  return u.bf;
}
// B fragment (32x16 bf16): source stored transposed baseT[n*ld + k].
// lane l: n = l&15; K = {h..h+7, 16+h..16+h+7}, h = 8*(l>=16).
static __device__ __forceinline__ v16bf load_b_bf16(const unsigned short* baseT,
                                                    int ld, int lane) {
  int n = lane & 15, h = (lane >> 4) << 3;
  const unsigned short* p = baseT + n * ld;
  FragU u;
  u.q[0] = *(const u32x4*)(p + h);
  u.q[1] = *(const u32x4*)(p + 16 + h);
  return u.bf;
}

// ---------------- k0: weight transposes to bf16 ----------------
__global__ void k0_prep(const float* W, const float* Wqkv, const float* Wo,
                        const float* Wf, unsigned short* WTb,
                        unsigned short* WqkvTb, unsigned short* WoTb,
                        unsigned short* WfTb) {
  int t = threadIdx.x;
  for (int i = t; i < 64 * 64; i += blockDim.x) {
    int n = i >> 6, k = i & 63;
    WTb[i]  = f2bf(W[k * 64 + n]);
    WoTb[i] = f2bf(Wo[k * 64 + n]);
    WfTb[i] = f2bf(Wf[k * 64 + n]);
  }
  for (int i = t; i < 192 * 64; i += blockDim.x) {
    int n = i >> 6, k = i & 63;
    WqkvTb[i] = f2bf(Wqkv[k * 192 + n]);
  }
}

// ---------------- k1: Wh = X@W (WMMA), WhT bf16, el/er ----------------
__global__ __launch_bounds__(128) void k1_gat_prep(const float* X,
                                                   const float* avec,
                                                   const unsigned short* WTb,
                                                   unsigned short* WhTb,
                                                   float* el, float* er) {
  __shared__ float Wh[4][16][64];
  int tid = threadIdx.x, w = tid >> 5, lane = tid & 31;
  int i0 = blockIdx.x * 64 + w * 16;
  v16bf a0 = load_a_f32(X + (size_t)i0 * 64, 64, lane);
  v16bf a1 = load_a_f32(X + (size_t)i0 * 64 + 32, 64, lane);
  int colL = lane & 15, hi8 = (lane >> 4) << 3;
#pragma unroll
  for (int t = 0; t < 4; ++t) {
    v8f c = {};
    c = wmma_bf16(a0, load_b_bf16(WTb + t * 16 * 64, 64, lane), c);
    c = wmma_bf16(a1, load_b_bf16(WTb + t * 16 * 64 + 32, 64, lane), c);
    int col = colL + t * 16;
#pragma unroll
    for (int k = 0; k < 8; ++k) Wh[w][k + hi8][col] = c[k];
  }
  __syncthreads();
  for (int c = lane; c < 64; c += 32) {
#pragma unroll
    for (int r = 0; r < 16; ++r)
      WhTb[(size_t)c * NND + i0 + r] = f2bf(Wh[w][r][c]);
  }
  if (lane < 16) {
    float s1 = 0.f, s2 = 0.f;
#pragma unroll
    for (int c = 0; c < 64; ++c) {
      float v = Wh[w][lane][c];
      s1 += v * avec[c];
      s2 += v * avec[64 + c];
    }
    el[i0 + lane] = s1;
    er[i0 + lane] = s2;
  }
}

// ---------------- k2: fused masked-softmax GAT aggregation ----------------
// e[i,j] = lrelu(el[i]+er[j]); mask = adj>0; h_gat = elu(softmax(e)@Wh)
__global__ __launch_bounds__(128) void k2_gat(const float* adj, const float* el,
                                              const float* er,
                                              const unsigned short* WhTb,
                                              float* hgat) {
  __shared__ float rowm[64], rowsi[64];
  int tid = threadIdx.x, w = tid >> 5, lane = tid & 31;
  int v = blockIdx.y;
  int iw = blockIdx.x * 64 + w * 16;
  const float* adjv = adj + (size_t)v * NND * NND;

  // phase 1: per-row online max/sum over masked e (coalesced adj stream)
  for (int rr = 0; rr < 16; ++rr) {
    int i = iw + rr;
    float eli = el[i];
    const float* arow = adjv + (size_t)i * NND;
    float m = -3.0e38f, s = 0.f;
    for (int j = lane; j < NND; j += 32) {
      float av = arow[j];
      if (av > 0.f) {
        float x = eli + er[j];
        x = x > 0.f ? x : LRELU_F * x;
        float mn = fmaxf(m, x);
        s = s * __expf(m - mn) + __expf(x - mn);
        m = mn;
      }
    }
#pragma unroll
    for (int off = 16; off > 0; off >>= 1) {
      float om = __shfl_xor(m, off, 32);
      float os = __shfl_xor(s, off, 32);
      float mn = fmaxf(m, om);
      float sn = 0.f;
      if (s > 0.f)  sn += s  * __expf(m - mn);
      if (os > 0.f) sn += os * __expf(om - mn);
      m = mn; s = sn;
    }
    if (lane == 0) { rowm[w * 16 + rr] = m; rowsi[w * 16 + rr] = 1.f / s; }
  }
  __syncthreads();

  // phase 2: P rebuilt per-lane in A-frag layout; accumulate P@Wh via WMMA
  int r = lane & 15, h = (lane >> 4) << 3, hi8 = h;
  int colL = lane & 15;
  int i = iw + r;
  float m = rowm[w * 16 + r];
  float si = rowsi[w * 16 + r];
  float eli = el[i];
  const float* arow = adjv + (size_t)i * NND;
  v8f acc[4] = {{}, {}, {}, {}};
  for (int jc = 0; jc < NND / 32; ++jc) {
    int jb = jc * 32;
    f32x4 A0 = __builtin_nontemporal_load((const f32x4*)(arow + jb + h));
    f32x4 A1 = __builtin_nontemporal_load((const f32x4*)(arow + jb + h + 4));
    f32x4 A2 = __builtin_nontemporal_load((const f32x4*)(arow + jb + 16 + h));
    f32x4 A3 = __builtin_nontemporal_load((const f32x4*)(arow + jb + 16 + h + 4));
    f32x4 E0 = *(const f32x4*)(er + jb + h);
    f32x4 E1 = *(const f32x4*)(er + jb + h + 4);
    f32x4 E2 = *(const f32x4*)(er + jb + 16 + h);
    f32x4 E3 = *(const f32x4*)(er + jb + 16 + h + 4);
    float av[16] = {A0[0],A0[1],A0[2],A0[3], A1[0],A1[1],A1[2],A1[3],
                    A2[0],A2[1],A2[2],A2[3], A3[0],A3[1],A3[2],A3[3]};
    float ev[16] = {E0[0],E0[1],E0[2],E0[3], E1[0],E1[1],E1[2],E1[3],
                    E2[0],E2[1],E2[2],E2[3], E3[0],E3[1],E3[2],E3[3]};
    FragU uA;
#pragma unroll
    for (int e = 0; e < 8; ++e) {
      float x0 = eli + ev[2 * e];
      x0 = x0 > 0.f ? x0 : LRELU_F * x0;
      float p0 = (av[2 * e] > 0.f) ? __expf(x0 - m) * si : 0.f;
      float x1 = eli + ev[2 * e + 1];
      x1 = x1 > 0.f ? x1 : LRELU_F * x1;
      float p1 = (av[2 * e + 1] > 0.f) ? __expf(x1 - m) * si : 0.f;
      uA.w[e] = pack_bf2(p0, p1);
    }
#pragma unroll
    for (int t = 0; t < 4; ++t)
      acc[t] = wmma_bf16(uA.bf, load_b_bf16(WhTb + t * 16 * NND + jb, NND, lane),
                         acc[t]);
  }
#pragma unroll
  for (int t = 0; t < 4; ++t) {
    int col = colL + t * 16;
#pragma unroll
    for (int k = 0; k < 8; ++k) {
      int ri = k + hi8;
      float x = acc[t][k];
      x = x > 0.f ? x : __expf(x) - 1.f;  // ELU
      hgat[((size_t)v * NND + iw + ri) * HD + col] = x;
    }
  }
}

// ---------------- k3a: qkv = h_gat@Wqkv + b (q pre-scaled by 1/8) ----------
__global__ __launch_bounds__(128) void k3a_qkv(const float* hgat,
                                               const unsigned short* WqkvTb,
                                               const float* bqkv,
                                               unsigned short* qb,
                                               unsigned short* kb,
                                               unsigned short* vTb) {
  int tid = threadIdx.x, lane = tid & 31;
  int g = blockIdx.x * 4 + (tid >> 5);
  int R0 = g * 16;
  int v = R0 / NND, i0 = R0 % NND;
  const float* base = hgat + ((size_t)v * NND + i0) * HD;
  v16bf a0 = load_a_f32(base, 64, lane);
  v16bf a1 = load_a_f32(base + 32, 64, lane);
  int colL = lane & 15, hi8 = (lane >> 4) << 3;
#pragma unroll
  for (int t = 0; t < 12; ++t) {
    v8f c = {};
    c = wmma_bf16(a0, load_b_bf16(WqkvTb + t * 16 * 64, 64, lane), c);
    c = wmma_bf16(a1, load_b_bf16(WqkvTb + t * 16 * 64 + 32, 64, lane), c);
    int col = colL + t * 16;
    float bias = bqkv[col];
#pragma unroll
    for (int k = 0; k < 8; ++k) {
      int iN = i0 + k + hi8;
      float val = c[k] + bias;
      if (col < 64)
        qb[((size_t)v * NND + iN) * 64 + col] = f2bf(val * 0.125f);
      else if (col < 128)
        kb[((size_t)v * NND + iN) * 64 + (col - 64)] = f2bf(val);
      else
        vTb[((size_t)v * 64 + (col - 128)) * NND + iN] = f2bf(val);
    }
  }
}

// ---------------- k3b: self-attn (S row-block in LDS) + Wo + fusion --------
__global__ __launch_bounds__(128) void k3b_attn_fuse(
    const unsigned short* qb, const unsigned short* kb,
    const unsigned short* vTb, const unsigned short* WoTb,
    const unsigned short* WfTb, const float* bo, const float* bfv,
    const float* hgat, float* hid, float* out) {
  extern __shared__ float sm[];
  float* S    = sm;                 // 16*3072 f32  (192 KB of 320 KB WGP LDS)
  float* red  = sm + 16 * NND;      // 128
  float* mS   = red + 128;          // 16
  float* sS   = mS + 16;            // 16
  float* part = sS + 16;            // 4*16*64
  float* Osm  = part + 4096;        // 16*64
  unsigned short* hfb = (unsigned short*)(Osm + 1024);  // 16*64 bf16

  int tid = threadIdx.x, w = tid >> 5, lane = tid & 31;
  int v = blockIdx.y;
  int i0 = blockIdx.x * 16;
  int colL = lane & 15, hi8 = (lane >> 4) << 3;
  const unsigned short* qv = qb + ((size_t)v * NND + i0) * 64;
  const unsigned short* kv = kb + (size_t)v * NND * 64;
  const unsigned short* vT = vTb + (size_t)v * 64 * NND;
  int jw = w * (NND / 4);

  // phase A: S = (q/8) @ k^T  into LDS
  {
    v16bf qa0 = load_a_bf16(qv, 64, lane);
    v16bf qa1 = load_a_bf16(qv + 32, 64, lane);
    for (int jt = 0; jt < (NND / 4) / 16; ++jt) {
      int j0 = jw + jt * 16;
      v8f c = {};
      c = wmma_bf16(qa0, load_b_bf16(kv + (size_t)j0 * 64, 64, lane), c);
      c = wmma_bf16(qa1, load_b_bf16(kv + (size_t)j0 * 64 + 32, 64, lane), c);
#pragma unroll
      for (int k = 0; k < 8; ++k) S[(k + hi8) * NND + j0 + colL] = c[k];
    }
  }
  __syncthreads();

  // phase B: row max then sum (8 threads/row)
  {
    int row = tid >> 3, seg = tid & 7;
    float m = -3.0e38f;
    for (int j = seg * 384; j < (seg + 1) * 384; ++j)
      m = fmaxf(m, S[row * NND + j]);
    red[tid] = m;
    __syncthreads();
    if (tid < 16) {
      float mm = red[tid * 8];
#pragma unroll
      for (int e = 1; e < 8; ++e) mm = fmaxf(mm, red[tid * 8 + e]);
      mS[tid] = mm;
    }
    __syncthreads();
    float mr = mS[row];
    float s = 0.f;
    for (int j = seg * 384; j < (seg + 1) * 384; ++j)
      s += __expf(S[row * NND + j] - mr);
    red[tid] = s;
    __syncthreads();
    if (tid < 16) {
      float ss = 0.f;
#pragma unroll
      for (int e = 0; e < 8; ++e) ss += red[tid * 8 + e];
      sS[tid] = 1.f / ss;
    }
    __syncthreads();
  }

  // phase C: O_partial = softmax(S) @ v  via WMMA
  {
    float mr = mS[colL];
    float si = sS[colL];
    int h = hi8;
    v8f o[4] = {{}, {}, {}, {}};
    for (int jc = 0; jc < (NND / 4) / 32; ++jc) {
      int jb = jw + jc * 32;
      const float* srow = S + colL * NND + jb;
      f32x4 A0 = *(const f32x4*)(srow + h);
      f32x4 A1 = *(const f32x4*)(srow + h + 4);
      f32x4 A2 = *(const f32x4*)(srow + 16 + h);
      f32x4 A3 = *(const f32x4*)(srow + 16 + h + 4);
      float xv[16] = {A0[0],A0[1],A0[2],A0[3], A1[0],A1[1],A1[2],A1[3],
                      A2[0],A2[1],A2[2],A2[3], A3[0],A3[1],A3[2],A3[3]};
      FragU uA;
#pragma unroll
      for (int e = 0; e < 8; ++e) {
        float p0 = __expf(xv[2 * e] - mr) * si;
        float p1 = __expf(xv[2 * e + 1] - mr) * si;
        uA.w[e] = pack_bf2(p0, p1);
      }
#pragma unroll
      for (int t = 0; t < 4; ++t)
        o[t] = wmma_bf16(uA.bf, load_b_bf16(vT + t * 16 * NND + jb, NND, lane),
                         o[t]);
    }
#pragma unroll
    for (int t = 0; t < 4; ++t)
#pragma unroll
      for (int k = 0; k < 8; ++k)
        part[w * 1024 + (k + hi8) * 64 + colL + t * 16] = o[t][k];
  }
  __syncthreads();

  // phase D: combine 4 wave partials
  for (int idx = tid; idx < 1024; idx += 128)
    Osm[idx] = part[idx] + part[1024 + idx] + part[2048 + idx] + part[3072 + idx];
  __syncthreads();
  if (w != 0) return;

  // phase E (wave 0): o = O@Wo + bo; h_fuse; w = sigmoid(h_fuse@Wf+bf);
  // hid = w*h_fuse; view_hidden_final = beta*hid + (1-beta)*h_fuse
  {
    v16bf oa0 = load_a_f32(Osm, 64, lane);
    v16bf oa1 = load_a_f32(Osm + 32, 64, lane);
    v8f hf[4];
#pragma unroll
    for (int t = 0; t < 4; ++t) {
      v8f c = {};
      c = wmma_bf16(oa0, load_b_bf16(WoTb + t * 16 * 64, 64, lane), c);
      c = wmma_bf16(oa1, load_b_bf16(WoTb + t * 16 * 64 + 32, 64, lane), c);
      int col = colL + t * 16;
      float bias = bo[col];
#pragma unroll
      for (int k = 0; k < 8; ++k) {
        int ri = k + hi8;
        float hg = hgat[((size_t)v * NND + i0 + ri) * HD + col];
        float x = ALPHA_F * (c[k] + bias) + (1.f - ALPHA_F) * hg;
        hf[t][k] = x;
        hfb[ri * 64 + col] = f2bf(x);
      }
    }
    asm volatile("s_wait_dscnt 0" ::: "memory");  // same-wave LDS RAW fence
    v16bf ha0 = load_a_bf16(hfb, 64, lane);
    v16bf ha1 = load_a_bf16(hfb + 32, 64, lane);
#pragma unroll
    for (int t = 0; t < 4; ++t) {
      v8f c = {};
      c = wmma_bf16(ha0, load_b_bf16(WfTb + t * 16 * 64, 64, lane), c);
      c = wmma_bf16(ha1, load_b_bf16(WfTb + t * 16 * 64 + 32, 64, lane), c);
      int col = colL + t * 16;
      float bias = bfv[col];
#pragma unroll
      for (int k = 0; k < 8; ++k) {
        int ri = k + hi8;
        float wv = 1.f / (1.f + __expf(-(c[k] + bias)));
        float hfx = hf[t][k];
        float hd = wv * hfx;
        size_t off = ((size_t)v * NND + i0 + ri) * HD + col;
        hid[off] = hd;
        out[65536 + off] = BETA_F * hd + (1.f - BETA_F) * hfx;
      }
    }
  }
}

// ---------------- k4: fusion_hidden = sum_v hid, rows 2048.. ----------------
__global__ void k4_fusion_sum(const float* hid, float* out) {
  int idx = blockIdx.x * blockDim.x + threadIdx.x;
  if (idx >= 1024 * 64) return;
  int r = idx >> 6, c = idx & 63;
  size_t base = ((size_t)(2048 + r)) * 64 + c;
  float s = 0.f;
#pragma unroll
  for (int v = 0; v < 4; ++v) s += hid[(size_t)v * NND * 64 + base];
  out[idx] = s;
}

extern "C" void kernel_launch(void* const* d_in, const int* in_sizes, int n_in,
                              void* d_out, int out_size, void* d_ws,
                              size_t ws_size, hipStream_t stream) {
  (void)in_sizes; (void)n_in; (void)out_size; (void)ws_size;
  const float* adj  = (const float*)d_in[0];
  const float* X    = (const float*)d_in[1];
  const float* W    = (const float*)d_in[2];
  const float* avec = (const float*)d_in[3];
  const float* Wqkv = (const float*)d_in[4];
  const float* bqkv = (const float*)d_in[5];
  const float* Wo   = (const float*)d_in[6];
  const float* bo   = (const float*)d_in[7];
  const float* Wf   = (const float*)d_in[8];
  const float* bfv  = (const float*)d_in[9];
  float* out = (float*)d_out;

  char* ws = (char*)d_ws;
  unsigned short* WTb    = (unsigned short*)(ws);
  unsigned short* WqkvTb = (unsigned short*)(ws + 8192);
  unsigned short* WoTb   = (unsigned short*)(ws + 32768);
  unsigned short* WfTb   = (unsigned short*)(ws + 40960);
  float*          el     = (float*)(ws + 49152);
  float*          er     = (float*)(ws + 61440);
  unsigned short* WhTb   = (unsigned short*)(ws + 73728);
  float*          hgat   = (float*)(ws + 466944);
  unsigned short* qb     = (unsigned short*)(ws + 3612672);
  unsigned short* kb     = (unsigned short*)(ws + 5185536);
  unsigned short* vTb    = (unsigned short*)(ws + 6758400);
  float*          hid    = (float*)(ws + 8331264);  // total ~11.5 MB

  k0_prep<<<1, 256, 0, stream>>>(W, Wqkv, Wo, Wf, WTb, WqkvTb, WoTb, WfTb);
  k1_gat_prep<<<48, 128, 0, stream>>>(X, avec, WTb, WhTb, el, er);
  k2_gat<<<dim3(48, 4), 128, 0, stream>>>(adj, el, er, WhTb, hgat);
  k3a_qkv<<<192, 128, 0, stream>>>(hgat, WqkvTb, bqkv, qb, kb, vTb);
  // dynamic LDS: 16*3072*4 + (128+16+16+4096+1024)*4 + 2048 = 219776 B (<320KB WGP)
  k3b_attn_fuse<<<dim3(192, 4), 128, 219776, stream>>>(
      qb, kb, vTb, WoTb, WfTb, bo, bfv, hgat, hid, out);
  k4_fusion_sum<<<(65536 + 255) / 256, 256, 0, stream>>>(hid, out);
}
